// ExpValCircuitGraphModel_30803505447185
// MI455X (gfx1250) — matrix operand order repaired and checked by reading, not compiled, not run on hardware
//
#include <hip/hip_runtime.h>

// ---------------------------------------------------------------------------
// ExpValCircuitGraphModel forward for MI455X (gfx1250, wave32, WMMA).
// All dense GEMMs use v_wmma_f32_16x16x32_bf16 (bf16 in, f32 accumulate).
// Weights are stored transposed (K-contiguous) at bf16-convert time so both
// A and B fragments are packed b128 loads in the dominant GEMMs. Segment ops
// use fp32 L2 atomics + order-preserving uint encoding for float atomic-max.
// Param pointers assumed in setup_inputs() dict insertion order.
// ---------------------------------------------------------------------------

typedef __bf16 bf16;
typedef __attribute__((ext_vector_type(16))) __bf16 v16bf;
typedef __attribute__((ext_vector_type(8)))  __bf16 v8bf;
typedef __attribute__((ext_vector_type(8)))  float  v8f;

namespace {
constexpr int NGR   = 128;            // graphs (B)
constexpr int NPGN  = 128;            // nodes per graph
constexpr int NTOT  = NGR * NPGN;     // 16384
constexpr int NEDGE = NTOT * 4;       // 65536
constexpr int NEE   = NEDGE + NTOT;   // edges incl self loops
constexpr int FIN   = 64;
constexpr int HCD   = 128;            // head channels
constexpr int D1    = HCD * 3;        // 384
constexpr int D2    = HCD * 2;        // 256
constexpr int KP1   = 64;
constexpr int KP2   = 32;
constexpr int NEV   = 4;
constexpr unsigned ENC_NEGINF = 0x007FFFFFu;       // encf(-inf)
constexpr float INV_SQRT_HC = 0.08838834764831845f;
}

#define DEV static __device__ __forceinline__

DEV bf16 f2bf(float f) {
  unsigned u = __builtin_bit_cast(unsigned, f);
  unsigned r = (u + 0x7fffu + ((u >> 16) & 1u)) >> 16;
  unsigned short s = (unsigned short)r;
  return __builtin_bit_cast(bf16, s);
}
DEV float bf2f(bf16 b) {
  unsigned short s = __builtin_bit_cast(unsigned short, b);
  unsigned u = ((unsigned)s) << 16;
  return __builtin_bit_cast(float, u);
}
DEV float bfu_lo(unsigned u) { return __builtin_bit_cast(float, u << 16); }
DEV float bfu_hi(unsigned u) { return __builtin_bit_cast(float, u & 0xffff0000u); }
// monotonic float<->uint mapping so atomicMax(uint) == float max
DEV unsigned encf(float f) {
  unsigned u = __builtin_bit_cast(unsigned, f);
  return (u & 0x80000000u) ? ~u : (u | 0x80000000u);
}
DEV float decf(unsigned e) {
  unsigned u = (e & 0x80000000u) ? (e & 0x7fffffffu) : ~e;
  return __builtin_bit_cast(float, u);
}
DEV float sigmoidf(float x) { return 1.0f / (1.0f + expf(-x)); }

// --- WMMA bf16 fragment loaders (contiguous k = h*16 + s in A and B, wave32) ---------------
// A 16x32 bf16: VGPR v, lane-half h: kbase = h*16 + v*2.
// Row-major A (k-stride 1): per lane the 16 elems are two contiguous 16B runs.
DEV v16bf load_frag_a_row(const bf16* base, long sr) {
  int lane = threadIdx.x & 31;
  int m  = lane & 15;
  int hf = lane >> 4;
  const bf16* p = base + (long)m * sr + hf * 16;
  v8bf lo = *(const v8bf*)(p);        // K = hf*16 + 0..7
  v8bf hi = *(const v8bf*)(p + 8);    // K = hf*16 + 8..15
  return __builtin_shufflevector(lo, hi, 0, 1, 2, 3, 4, 5, 6, 7,
                                 8, 9, 10, 11, 12, 13, 14, 15);
}
// B 32x16 bf16: VGPR v, lane-half h: kbase = h*16 + v*2.
// Generic (bsn==1 activation case): per-lane u16 loads, coalesced across lanes.
DEV v16bf load_frag_b_gen(const bf16* base, long sk, long sn) {
  int lane = threadIdx.x & 31;
  int n  = lane & 15;
  int hf = lane >> 4;
  const bf16* p = base + (long)n * sn;
  v16bf r;
#pragma unroll
  for (int v = 0; v < 8; ++v) {
    int kb = (hf << 4) + (v << 1);
    r[2 * v]     = p[(long)kb * sk];
    r[2 * v + 1] = p[(long)(kb + 1) * sk];
  }
  return r;
}
// B with contiguous K (bsk==1: transposed weights / K^T): two b128 loads/lane.
DEV v16bf load_frag_b_k1(const bf16* base, long sn) {
  int lane = threadIdx.x & 31;
  int n  = lane & 15;
  int hf = lane >> 4;
  const bf16* p = base + (long)n * sn + hf * 16;
  v8bf lo = *(const v8bf*)(p);        // K = hf*16 + 0..7
  v8bf hi = *(const v8bf*)(p + 8);    // K = hf*16 + 8..15
  return __builtin_shufflevector(lo, hi, 0, 1, 2, 3, 4, 5, 6, 7,
                                 8, 9, 10, 11, 12, 13, 14, 15);
}

DEV void edge_sd(int id, const int* src, const int* dst, int& s, int& d) {
  if (id < NEDGE) { s = src[id]; d = dst[id]; }
  else            { s = id - NEDGE; d = s; }
}

// ------------------------------ kernels ------------------------------------

__global__ void k_cvt_bf(const float* __restrict__ in, bf16* __restrict__ out, int n) {
  int t = blockIdx.x * 256 + threadIdx.x;
  if (t < n) out[t] = f2bf(in[t]);
}
// weight convert + transpose: in = W[K][N] row-major -> out = Wt[N][K]
__global__ void k_cvt_bf_t(const float* __restrict__ in, bf16* __restrict__ out,
                           int K, int N) {
  int t = blockIdx.x * 256 + threadIdx.x;
  if (t >= K * N) return;
  int n = t / K, k = t - n * K;
  out[t] = f2bf(in[(long)k * N + n]);
}
__global__ void k_fill_f32(float* p, float v, int n) {
  int t = blockIdx.x * 256 + threadIdx.x;
  if (t < n) p[t] = v;
}
__global__ void k_fill_u32(unsigned* p, unsigned v, int n) {
  int t = blockIdx.x * 256 + threadIdx.x;
  if (t < n) p[t] = v;
}
__global__ void k_enc_to_bf(const unsigned* __restrict__ in, bf16* __restrict__ out, int n) {
  int t = blockIdx.x * 256 + threadIdx.x;
  if (t < n) out[t] = f2bf(decf(in[t]));
}

// Strided batched GEMM: C[m,n] = sum_k A(m,k)*B(k,n) (+bias[n]).
// A is row-major (k-stride 1): packed b128 fragment loads.
// BK1: B has k-stride 1 (transposed weights) -> packed b128 fragment loads.
// OM:  0 = store f32, 1 = store bf16 (branch-free epilogue).
// One wave computes one 16x16 tile; block = 4 waves; grid.z = batch.
template <bool BK1, int OM>
__global__ __launch_bounds__(128)
void k_gemm_bf16(const bf16* __restrict__ A, const bf16* __restrict__ B,
                 const float* __restrict__ bias, void* __restrict__ Cout,
                 int M, int N, int K,
                 long asr, long abat,
                 long bsk, long bsn, long bbat,
                 long csr, long cbat) {
  int mtile = blockIdx.y * 4 + (threadIdx.x >> 5);
  if (mtile * 16 >= M) return;
  int ntile = blockIdx.x;
  long z = blockIdx.z;
  const bf16* Ab = A + z * abat + (long)mtile * 16 * asr;
  const bf16* Bb = B + z * bbat + (long)ntile * 16 * bsn;
  int lane = threadIdx.x & 31;
  int n  = ntile * 16 + (lane & 15);
  int hf = lane >> 4;
  v8f acc = {};
  if (bias) {
    float bv = bias[n];
#pragma unroll
    for (int r = 0; r < 8; ++r) acc[r] = bv;
  }
  for (int kb = 0; kb < K; kb += 32) {
    v16bf af = load_frag_a_row(Ab + kb, asr);
    v16bf bf_;
    if constexpr (BK1) bf_ = load_frag_b_k1(Bb + kb, bsn);
    else               bf_ = load_frag_b_gen(Bb + (long)kb * bsk, bsk, bsn);
    acc = __builtin_amdgcn_wmma_f32_16x16x32_bf16(false, af, false, bf_,
                                                  (short)0, acc, false, false);
  }
#pragma unroll
  for (int r = 0; r < 8; ++r) {
    int m = mtile * 16 + r + hf * 8;
    long o = z * cbat + (long)m * csr + n;
    if constexpr (OM == 0) ((float*)Cout)[o] = acc[r];
    else                   ((bf16*)Cout)[o]  = f2bf(acc[r]);
  }
}

// t1 sparse attention: alpha[e,h] = <q[dst],k[src]>/sqrt(HC); seg-max by dst
__global__ void k_t1_alpha(const bf16* __restrict__ q, const bf16* __restrict__ k,
                           const int* __restrict__ src, const int* __restrict__ dst,
                           float* __restrict__ alpha, unsigned* __restrict__ menc) {
  int t = blockIdx.x * 256 + threadIdx.x;
  if (t >= NEDGE * 3) return;
  int e = t / 3, h = t - e * 3;
  int s = src[e], d = dst[e];
  const unsigned* qp = (const unsigned*)(q + (long)d * D1 + h * HCD);
  const unsigned* kp = (const unsigned*)(k + (long)s * D1 + h * HCD);
  float acc = 0.f;
  for (int c = 0; c < HCD / 2; ++c) {
    unsigned qa = qp[c], ka = kp[c];
    acc += bfu_lo(qa) * bfu_lo(ka) + bfu_hi(qa) * bfu_hi(ka);
  }
  acc *= INV_SQRT_HC;
  alpha[t] = acc;
  atomicMax(&menc[d * 3 + h], encf(acc));
}

__global__ void k_t1_exp(const int* __restrict__ dst, const unsigned* __restrict__ menc,
                         float* __restrict__ alpha, float* __restrict__ z) {
  int t = blockIdx.x * 256 + threadIdx.x;
  if (t >= NEDGE * 3) return;
  int e = t / 3, h = t - e * 3;
  int d = dst[e];
  float m = decf(menc[d * 3 + h]);
  if (m < -3.0e38f) m = 0.f;                   // where(isfinite(m), m, 0)
  float ev = expf(alpha[t] - m);
  alpha[t] = ev;
  atomicAdd(&z[d * 3 + h], ev);
}

__global__ void k_t1_msg(const float* __restrict__ alpha, const float* __restrict__ z,
                         const bf16* __restrict__ v, const int* __restrict__ src,
                         const int* __restrict__ dst, float* __restrict__ x1) {
  int t = blockIdx.x * 256 + threadIdx.x;
  if (t >= NEDGE * D1) return;
  int e = t / D1, j = t - e * D1;
  int h = j >> 7;
  int s = src[e], d = dst[e];
  float at = alpha[e * 3 + h] / fmaxf(z[d * 3 + h], 1e-16f);
  atomicAdd(&x1[(long)d * D1 + j], at * bf2f(v[(long)s * D1 + j]));
}

// pool1: seg-max of x over incoming edges (self loop = init)
__global__ void k_xmax_init(const float* __restrict__ x1, unsigned* __restrict__ xe, int n) {
  int t = blockIdx.x * 256 + threadIdx.x;
  if (t < n) xe[t] = encf(x1[t]);
}
__global__ void k_xmax_edge(const float* __restrict__ x1, const int* __restrict__ src,
                            const int* __restrict__ dst, unsigned* __restrict__ xe) {
  int t = blockIdx.x * 256 + threadIdx.x;
  if (t >= NEDGE * D1) return;
  int e = t / D1, j = t - e * D1;
  atomicMax(&xe[(long)dst[e] * D1 + j], encf(x1[(long)src[e] * D1 + j]));
}

__global__ void k_matvec(const float* __restrict__ X, const float* __restrict__ w,
                         const float* __restrict__ bias, float* __restrict__ out,
                         int M, int K) {
  int i = blockIdx.x * 256 + threadIdx.x;
  if (i >= M) return;
  const float* x = X + (long)i * K;
  float acc = bias ? bias[0] : 0.f;
  for (int c = 0; c < K; ++c) acc += x[c] * w[c];
  out[i] = acc;
}

__global__ void k_fit(const float* __restrict__ X, const float* __restrict__ w,
                      const float* __restrict__ bias, const float* __restrict__ extra,
                      float* __restrict__ out, int M, int K) {
  int i = blockIdx.x * 256 + threadIdx.x;
  if (i >= M) return;
  const float* x = X + (long)i * K;
  float acc = bias[0] + extra[i];
  for (int c = 0; c < K; ++c) acc += x[c] * w[c];
  out[i] = sigmoidf(acc);
}

__global__ void k_p1_score(const int* __restrict__ src, const int* __restrict__ dst,
                           const float* __restrict__ tq, const float* __restrict__ tj,
                           const float* __restrict__ ab, float* __restrict__ score,
                           unsigned* __restrict__ menc) {
  int id = blockIdx.x * 256 + threadIdx.x;
  if (id >= NEE) return;
  int s, d; edge_sd(id, src, dst, s, d);
  float v = tq[d] + tj[s] + ab[0];
  v = v > 0.f ? v : 0.2f * v;                  // leaky_relu 0.2
  score[id] = v;
  atomicMax(&menc[d], encf(v));
}
__global__ void k_p1_exp(const int* __restrict__ src, const int* __restrict__ dst,
                         const unsigned* __restrict__ menc, float* __restrict__ score,
                         float* __restrict__ z) {
  int id = blockIdx.x * 256 + threadIdx.x;
  if (id >= NEE) return;
  int s, d; edge_sd(id, src, dst, s, d);
  float ev = expf(score[id] - decf(menc[d]));
  score[id] = ev;
  atomicAdd(&z[d], ev);
}
__global__ void k_p1_norm(const int* __restrict__ src, const int* __restrict__ dst,
                          const float* __restrict__ z, float* __restrict__ score) {
  int id = blockIdx.x * 256 + threadIdx.x;
  if (id >= NEE) return;
  int s, d; edge_sd(id, src, dst, s, d);
  score[id] = score[id] / fmaxf(z[d], 1e-16f);
}
__global__ void k_p1_xp(const int* __restrict__ src, const int* __restrict__ dst,
                        const float* __restrict__ score, const float* __restrict__ x1,
                        float* __restrict__ xp) {
  int t = blockIdx.x * 256 + threadIdx.x;
  if (t >= NEE * D1) return;
  int id = t / D1, j = t - id * D1;
  int s, d; edge_sd(id, src, dst, s, d);
  atomicAdd(&xp[(long)d * D1 + j], score[id] * x1[(long)s * D1 + j]);
}
__global__ void k_p1_agg(const int* __restrict__ src, const int* __restrict__ dst,
                         const float* __restrict__ a, const float* __restrict__ bb,
                         float* __restrict__ agg) {
  int id = blockIdx.x * 256 + threadIdx.x;
  if (id >= NEE) return;
  int s, d; edge_sd(id, src, dst, s, d);
  atomicAdd(&agg[d], a[s] - bb[d]);
}

// stable descending top-k by rank counting (one block per graph)
__global__ void k_topk(const float* __restrict__ fit, int npg, int kk,
                       int* __restrict__ idx, float* __restrict__ fsel) {
  __shared__ float sv[128];
  int b = blockIdx.x, i = threadIdx.x;
  sv[i] = fit[b * npg + i];
  __syncthreads();
  float v = sv[i];
  int rank = 0;
  for (int j = 0; j < npg; ++j) {
    float w = sv[j];
    rank += (w > v) || (w == v && j < i);
  }
  if (rank < kk) { idx[b * kk + rank] = i; fsel[b * kk + rank] = v; }
}

__global__ void k_gather_xsel1(const float* __restrict__ xp, const int* __restrict__ idx,
                               const float* __restrict__ fsel, bf16* __restrict__ out) {
  int t = blockIdx.x * 256 + threadIdx.x;
  if (t >= NGR * KP1 * D1) return;
  int b = t / (KP1 * D1), r = t - b * (KP1 * D1);
  int kk = r / D1, j = r - kk * D1;
  int node = b * NPGN + idx[b * KP1 + kk];
  out[t] = f2bf(xp[(long)node * D1 + j] * fsel[b * KP1 + kk]);
}

__global__ void k_p1_SA(const int* __restrict__ src, const int* __restrict__ dst,
                        const float* __restrict__ score, float* __restrict__ S,
                        float* __restrict__ A) {
  int id = blockIdx.x * 256 + threadIdx.x;
  if (id >= NEE) return;
  int s, d; edge_sd(id, src, dst, s, d);
  int g = s / NPGN;
  long o = (long)g * NPGN * NPGN + (s - g * NPGN) * NPGN + (d % NPGN);
  atomicAdd(&S[o], score[id]);
  atomicAdd(&A[o], 1.0f);
}
// SselT[b][l][j] = S[b][j][ idx[b][l] ]   (K-contiguous for WMMA B operand)
__global__ void k_gather_sselT(const float* __restrict__ S, const int* __restrict__ idx,
                               bf16* __restrict__ out) {
  int t = blockIdx.x * 256 + threadIdx.x;
  if (t >= NGR * KP1 * NPGN) return;
  int b = t / (KP1 * NPGN), r = t - b * (KP1 * NPGN);
  int l = r / NPGN, j = r - l * NPGN;
  out[t] = f2bf(S[(long)b * NPGN * NPGN + j * NPGN + idx[b * KP1 + l]]);
}
__global__ void k_adj(const float* __restrict__ A2, unsigned char* __restrict__ adj) {
  int t = blockIdx.x * 256 + threadIdx.x;
  if (t >= NGR * KP1 * KP1) return;
  int r = t % (KP1 * KP1);
  int i = r / KP1, j = r - i * KP1;
  adj[t] = (i != j && A2[t] > 0.f) ? 1 : 0;
}

// t2 dense attention: logits[b,h] = Q_bh @ K_bh^T / sqrt(HC)  (WMMA)
__global__ __launch_bounds__(128)
void k_attn_logits_t2(const bf16* __restrict__ q2, const bf16* __restrict__ k2,
                      float* __restrict__ logits) {
  int z = blockIdx.z, b = z >> 1, h = z & 1;
  int ntile = blockIdx.x;
  int mtile = threadIdx.x >> 5;
  const bf16* Qb = q2 + (long)b * KP1 * D2 + h * HCD + (long)mtile * 16 * D2;
  const bf16* Kb = k2 + (long)b * KP1 * D2 + h * HCD + (long)ntile * 16 * D2;
  v8f acc = {};
  for (int kb = 0; kb < HCD; kb += 32) {
    v16bf a  = load_frag_a_row(Qb + kb, D2);
    v16bf bb = load_frag_b_k1(Kb + kb, D2);   // K^T: (k,n) -> Kb[n*D2 + k]
    acc = __builtin_amdgcn_wmma_f32_16x16x32_bf16(false, a, false, bb,
                                                  (short)0, acc, false, false);
  }
  int lane = threadIdx.x & 31;
  int n = ntile * 16 + (lane & 15), hf = lane >> 4;
#pragma unroll
  for (int r = 0; r < 8; ++r) {
    int m = mtile * 16 + r + hf * 8;
    logits[(long)z * KP1 * KP1 + m * KP1 + n] = acc[r] * INV_SQRT_HC;
  }
}

__global__ void k_t2_softmax(const unsigned char* __restrict__ adj, float* __restrict__ lg) {
  int t = blockIdx.x * 256 + threadIdx.x;
  if (t >= 2 * NGR * KP1) return;
  int z = t >> 6, i = t & 63, b = z >> 1;
  float* row = lg + (long)z * KP1 * KP1 + i * KP1;
  const unsigned char* ad = adj + (long)b * KP1 * KP1;
  float mx = -1e30f;
  for (int j = 0; j < KP1; ++j) {
    float v = ad[j * KP1 + i] ? row[j] : -1e30f;   // mask = adj^T
    mx = fmaxf(mx, v);
  }
  float sum = 0.f;
  for (int j = 0; j < KP1; ++j) {
    float e = ad[j * KP1 + i] ? expf(row[j] - mx) : 0.f;
    row[j] = e;
    sum += e;
  }
  float inv = 1.f / fmaxf(sum, 1e-16f);
  for (int j = 0; j < KP1; ++j) row[j] *= inv;
}

// out[b,i,h*HC+c] += sum_j attn[z,i,j] * v2[b,j,h*HC+c]  (WMMA, adds into x2)
__global__ __launch_bounds__(128)
void k_attn_apply_t2(const bf16* __restrict__ attn, const bf16* __restrict__ v2,
                     float* __restrict__ x2) {
  int z = blockIdx.z, b = z >> 1, h = z & 1;
  int ntile = blockIdx.x;
  int mtile = threadIdx.x >> 5;
  const bf16* Ab = attn + (long)z * KP1 * KP1 + (long)mtile * 16 * KP1;
  const bf16* Bb = v2 + (long)b * KP1 * D2 + h * HCD + ntile * 16;
  v8f acc = {};
  for (int kb = 0; kb < KP1; kb += 32) {
    v16bf a  = load_frag_a_row(Ab + kb, KP1);
    v16bf bb = load_frag_b_gen(Bb + (long)kb * D2, D2, 1);
    acc = __builtin_amdgcn_wmma_f32_16x16x32_bf16(false, a, false, bb,
                                                  (short)0, acc, false, false);
  }
  int lane = threadIdx.x & 31;
  int n = ntile * 16 + (lane & 15), hf = lane >> 4;
#pragma unroll
  for (int r = 0; r < 8; ++r) {
    int m = mtile * 16 + r + hf * 8;
    long o = ((long)b * KP1 + m) * D2 + h * HCD + n;
    x2[o] += acc[r];
  }
}

// pool2: masked row-max of x2 (mask_in = (adj|I)^T), direct dense loops
__global__ void k_p2_xmax(const float* __restrict__ x2, const unsigned char* __restrict__ adj,
                          bf16* __restrict__ out) {
  int t = blockIdx.x * 256 + threadIdx.x;
  if (t >= NGR * KP1 * D2) return;
  int b = t / (KP1 * D2), r = t - b * (KP1 * D2);
  int i = r / D2, dch = r - i * D2;
  const float* xb = x2 + (long)b * KP1 * D2;
  const unsigned char* ad = adj + (long)b * KP1 * KP1;
  float m = xb[i * D2 + dch];                       // self loop
  for (int j = 0; j < KP1; ++j)
    if (ad[j * KP1 + i]) m = fmaxf(m, xb[j * D2 + dch]);
  out[t] = f2bf(m);
}

__global__ void k_p2_attn(const float* __restrict__ si, const float* __restrict__ tj,
                          const unsigned char* __restrict__ adj, const float* __restrict__ ab,
                          float* __restrict__ attn) {
  int t = blockIdx.x * 256 + threadIdx.x;
  if (t >= NGR * KP1) return;
  int b = t >> 6, i = t & 63;
  float siv = si[t], ab0 = ab[0];
  const float* tjb = tj + b * KP1;
  const unsigned char* ad = adj + (long)b * KP1 * KP1;
  float mx = -1e30f;
  for (int j = 0; j < KP1; ++j) {
    bool msk = (i == j) || ad[j * KP1 + i];
    float v = siv + tjb[j] + ab0;
    v = v > 0.f ? v : 0.2f * v;
    mx = fmaxf(mx, msk ? v : -1e30f);
  }
  float sum = 0.f;
  float* out = attn + (long)t * KP1;
  for (int j = 0; j < KP1; ++j) {
    bool msk = (i == j) || ad[j * KP1 + i];
    float v = siv + tjb[j] + ab0;
    v = v > 0.f ? v : 0.2f * v;
    float e = msk ? expf(v - mx) : 0.f;
    out[j] = e;
    sum += e;
  }
  float inv = 1.f / fmaxf(sum, 1e-16f);
  for (int j = 0; j < KP1; ++j) out[j] *= inv;
}

__global__ void k_p2_agg(const unsigned char* __restrict__ adj, const float* __restrict__ a,
                         const float* __restrict__ bb, float* __restrict__ agg) {
  int t = blockIdx.x * 256 + threadIdx.x;
  if (t >= NGR * KP1) return;
  int b = t >> 6, i = t & 63;
  const unsigned char* ad = adj + (long)b * KP1 * KP1;
  float sa = 0.f, cnt = 0.f;
  for (int j = 0; j < KP1; ++j) {
    bool msk = (i == j) || ad[j * KP1 + i];
    if (msk) { sa += a[b * KP1 + j]; cnt += 1.f; }
  }
  agg[t] = sa - cnt * bb[t];
}

__global__ void k_pool_mean(const float* __restrict__ xp2, const int* __restrict__ idx,
                            const float* __restrict__ fsel, float* __restrict__ g) {
  int t = blockIdx.x * 256 + threadIdx.x;
  if (t >= NGR * D2) return;
  int b = t / D2, dch = t - b * D2;
  float acc = 0.f;
  for (int k = 0; k < KP2; ++k) {
    int node = b * KP1 + idx[b * KP2 + k];
    acc += xp2[(long)node * D2 + dch] * fsel[b * KP2 + k];
  }
  g[t] = acc * (1.0f / KP2);
}

__global__ void k_head1(const float* __restrict__ g, const float* __restrict__ expv,
                        const float* __restrict__ depth, const float* __restrict__ w,
                        const float* __restrict__ bias, float* __restrict__ h) {
  int t = blockIdx.x * 256 + threadIdx.x;
  if (t >= NGR * HCD) return;
  int b = t >> 7, j = t & 127;
  float acc = bias[j];
  const float* gb = g + b * D2;
  for (int c = 0; c < D2; ++c) acc += gb[c] * w[c * HCD + j];
  for (int c = 0; c < NEV; ++c) acc += expv[b * NEV + c] * w[(D2 + c) * HCD + j];
  acc += depth[b] * w[(D2 + NEV) * HCD + j];
  h[t] = acc;
}
__global__ void k_head2(const float* __restrict__ h, const float* __restrict__ w,
                        const float* __restrict__ bias, float* __restrict__ out) {
  int t = blockIdx.x * 256 + threadIdx.x;
  if (t >= NGR * NEV) return;
  int b = t >> 2, o = t & 3;
  float acc = bias[o];
  for (int j = 0; j < HCD; ++j) acc += h[b * HCD + j] * w[j * NEV + o];
  out[t] = acc;
}

// ------------------------------- host --------------------------------------

static inline unsigned blks(long n) { return (unsigned)((n + 255) / 256); }

static void gemm(hipStream_t st, const bf16* A, const bf16* Bm, const float* bias,
                 float* C, bf16* Cbf, int M, int N, int K,
                 long asr, long abat,
                 long bsk, long bsn, long bbat,
                 long csr, long cbat, int nbat) {
  dim3 g((N + 15) / 16, (M / 16 + 3) / 4, nbat);
  bool bk1 = (bsk == 1);
  if (Cbf) {
    if (bk1) k_gemm_bf16<true, 1><<<g, 128, 0, st>>>(A, Bm, bias, Cbf, M, N, K,
                         asr, abat, bsk, bsn, bbat, csr, cbat);
    else     k_gemm_bf16<false, 1><<<g, 128, 0, st>>>(A, Bm, bias, Cbf, M, N, K,
                         asr, abat, bsk, bsn, bbat, csr, cbat);
  } else {
    if (bk1) k_gemm_bf16<true, 0><<<g, 128, 0, st>>>(A, Bm, bias, C, M, N, K,
                         asr, abat, bsk, bsn, bbat, csr, cbat);
    else     k_gemm_bf16<false, 0><<<g, 128, 0, st>>>(A, Bm, bias, C, M, N, K,
                         asr, abat, bsk, bsn, bbat, csr, cbat);
  }
}

extern "C" void kernel_launch(void* const* d_in, const int* in_sizes, int n_in,
                              void* d_out, int out_size, void* d_ws, size_t ws_size,
                              hipStream_t stream) {
  (void)in_sizes; (void)n_in; (void)out_size; (void)ws_size;
  // inputs in setup_inputs() dict order; params flattened in insertion order
  const float* expv  = (const float*)d_in[0];
  const float* depth = (const float*)d_in[2];
  const float* nodes = (const float*)d_in[3];
  const int*   eidx  = (const int*)d_in[4];
  const int* src = eidx;
  const int* dst = eidx + NEDGE;
  const float* t1_wq = (const float*)d_in[6];
  const float* t1_wk = (const float*)d_in[7];
  const float* t1_wv = (const float*)d_in[8];
  const float* t1_ws = (const float*)d_in[9];
  const float* t1_bq = (const float*)d_in[10];
  const float* t1_bk = (const float*)d_in[11];
  const float* t1_bv = (const float*)d_in[12];
  const float* t1_bs = (const float*)d_in[13];
  const float* t2_wq = (const float*)d_in[14];
  const float* t2_wk = (const float*)d_in[15];
  const float* t2_wv = (const float*)d_in[16];
  const float* t2_ws = (const float*)d_in[17];
  const float* t2_bq = (const float*)d_in[18];
  const float* t2_bk = (const float*)d_in[19];
  const float* t2_bv = (const float*)d_in[20];
  const float* t2_bs = (const float*)d_in[21];
  const float* p1_lw = (const float*)d_in[22];
  const float* p1_lb = (const float*)d_in[23];
  const float* p1_aw = (const float*)d_in[24];
  const float* p1_ab = (const float*)d_in[25];
  const float* p1_le1w = (const float*)d_in[26];
  const float* p1_le1b = (const float*)d_in[27];
  const float* p1_le2w = (const float*)d_in[28];
  const float* p1_le3w = (const float*)d_in[29];
  const float* p1_le3b = (const float*)d_in[30];
  const float* p2_lw = (const float*)d_in[31];
  const float* p2_lb = (const float*)d_in[32];
  const float* p2_aw = (const float*)d_in[33];
  const float* p2_ab = (const float*)d_in[34];
  const float* p2_le1w = (const float*)d_in[35];
  const float* p2_le1b = (const float*)d_in[36];
  const float* p2_le2w = (const float*)d_in[37];
  const float* p2_le3w = (const float*)d_in[38];
  const float* p2_le3b = (const float*)d_in[39];
  const float* b1_w = (const float*)d_in[40];
  const float* b1_b = (const float*)d_in[41];
  const float* b2_w = (const float*)d_in[42];
  const float* b2_b = (const float*)d_in[43];
  float* out = (float*)d_out;

  // workspace bump allocator (256B aligned)
  char* base = (char*)d_ws;
  size_t off = 0;
  auto alloc = [&](size_t elems, size_t esz) -> void* {
    void* p = base + off;
    off += (elems * esz + 255) & ~(size_t)255;
    return p;
  };
  bf16* nodes_bf = (bf16*)alloc((size_t)NTOT * FIN, 2);
  bf16* wq1b = (bf16*)alloc((size_t)FIN * D1, 2);     // transposed [N][K]
  bf16* wk1b = (bf16*)alloc((size_t)FIN * D1, 2);
  bf16* wv1b = (bf16*)alloc((size_t)FIN * D1, 2);
  bf16* ws1b = (bf16*)alloc((size_t)FIN * D1, 2);
  bf16* p1lwb = (bf16*)alloc((size_t)D1 * D1, 2);
  bf16* wq2b = (bf16*)alloc((size_t)D1 * D2, 2);
  bf16* wk2b = (bf16*)alloc((size_t)D1 * D2, 2);
  bf16* wv2b = (bf16*)alloc((size_t)D1 * D2, 2);
  bf16* ws2b = (bf16*)alloc((size_t)D1 * D2, 2);
  bf16* p2lwb = (bf16*)alloc((size_t)D2 * D2, 2);
  bf16* q1b = (bf16*)alloc((size_t)NTOT * D1, 2);
  bf16* k1b = (bf16*)alloc((size_t)NTOT * D1, 2);
  bf16* v1b = (bf16*)alloc((size_t)NTOT * D1, 2);
  float* x1 = (float*)alloc((size_t)NTOT * D1, 4);
  float* alpha = (float*)alloc((size_t)NEDGE * 3, 4);
  unsigned* menc1 = (unsigned*)alloc((size_t)NTOT * 3, 4);
  float* z1 = (float*)alloc((size_t)NTOT * 3, 4);
  unsigned* xme = (unsigned*)alloc((size_t)NTOT * D1, 4);
  bf16* xmaxb = (bf16*)alloc((size_t)NTOT * D1, 2);
  float* xq1 = (float*)alloc((size_t)NTOT * D1, 4);
  float* tqv = (float*)alloc(NTOT, 4);
  float* tjv = (float*)alloc(NTOT, 4);
  float* score = (float*)alloc(NEE, 4);
  unsigned* menc2 = (unsigned*)alloc(NTOT, 4);
  float* z2 = (float*)alloc(NTOT, 4);
  float* xp1 = (float*)alloc((size_t)NTOT * D1, 4);
  float* a1 = (float*)alloc(NTOT, 4);
  float* bb1 = (float*)alloc(NTOT, 4);
  float* agg1 = (float*)alloc(NTOT, 4);
  float* fit1 = (float*)alloc(NTOT, 4);
  int* idx1 = (int*)alloc((size_t)NGR * KP1, 4);
  float* fsel1 = (float*)alloc((size_t)NGR * KP1, 4);
  bf16* xsel1b = (bf16*)alloc((size_t)NGR * KP1 * D1, 2);
  float* Sbuf = (float*)alloc((size_t)NGR * NPGN * NPGN, 4);
  float* Abuf = (float*)alloc((size_t)NGR * NPGN * NPGN, 4);
  bf16* Abf = (bf16*)alloc((size_t)NGR * NPGN * NPGN, 2);
  bf16* sselT = (bf16*)alloc((size_t)NGR * KP1 * NPGN, 2);  // [b][l][j]
  bf16* Tbf = (bf16*)alloc((size_t)NGR * NPGN * KP1, 2);    // [b][i][l]
  float* A2 = (float*)alloc((size_t)NGR * KP1 * KP1, 4);
  unsigned char* adj = (unsigned char*)alloc((size_t)NGR * KP1 * KP1, 1);
  bf16* q2b = (bf16*)alloc((size_t)NGR * KP1 * D2, 2);
  bf16* k2b = (bf16*)alloc((size_t)NGR * KP1 * D2, 2);
  bf16* v2b = (bf16*)alloc((size_t)NGR * KP1 * D2, 2);
  float* x2 = (float*)alloc((size_t)NGR * KP1 * D2, 4);
  bf16* x2b = (bf16*)alloc((size_t)NGR * KP1 * D2, 2);
  float* logits = (float*)alloc((size_t)NGR * 2 * KP1 * KP1, 4);
  bf16* attnb = (bf16*)alloc((size_t)NGR * 2 * KP1 * KP1, 2);
  bf16* xmax2b = (bf16*)alloc((size_t)NGR * KP1 * D2, 2);
  float* xq2 = (float*)alloc((size_t)NGR * KP1 * D2, 4);
  float* si = (float*)alloc((size_t)NGR * KP1, 4);
  float* tj2 = (float*)alloc((size_t)NGR * KP1, 4);
  float* attn2 = (float*)alloc((size_t)NGR * KP1 * KP1, 4);
  bf16* attn2b = (bf16*)alloc((size_t)NGR * KP1 * KP1, 2);
  float* xp2 = (float*)alloc((size_t)NGR * KP1 * D2, 4);
  float* a2 = (float*)alloc((size_t)NGR * KP1, 4);
  float* bb2 = (float*)alloc((size_t)NGR * KP1, 4);
  float* agg2 = (float*)alloc((size_t)NGR * KP1, 4);
  float* fit2 = (float*)alloc((size_t)NGR * KP1, 4);
  int* idx2 = (int*)alloc((size_t)NGR * KP2, 4);
  float* fsel2 = (float*)alloc((size_t)NGR * KP2, 4);
  float* gmean = (float*)alloc((size_t)NGR * D2, 4);
  float* hbuf = (float*)alloc((size_t)NGR * HCD, 4);

#define CVT(src_, dst_, n_) k_cvt_bf<<<blks(n_), 256, 0, stream>>>((src_), (dst_), (int)(n_))
#define CVTT(src_, dst_, K_, N_) \
  k_cvt_bf_t<<<blks((long)(K_) * (N_)), 256, 0, stream>>>((src_), (dst_), (K_), (N_))
#define FILLF(p_, v_, n_) k_fill_f32<<<blks(n_), 256, 0, stream>>>((p_), (v_), (int)(n_))
#define FILLU(p_, v_, n_) k_fill_u32<<<blks(n_), 256, 0, stream>>>((p_), (v_), (int)(n_))

  // --- bf16 conversions (weights transposed to K-contiguous) ---
  CVT(nodes, nodes_bf, NTOT * FIN);
  CVTT(t1_wq, wq1b, FIN, D1); CVTT(t1_wk, wk1b, FIN, D1);
  CVTT(t1_wv, wv1b, FIN, D1); CVTT(t1_ws, ws1b, FIN, D1);
  CVTT(p1_lw, p1lwb, D1, D1);
  CVTT(t2_wq, wq2b, D1, D2); CVTT(t2_wk, wk2b, D1, D2);
  CVTT(t2_wv, wv2b, D1, D2); CVTT(t2_ws, ws2b, D1, D2);
  CVTT(p2_lw, p2lwb, D2, D2);

  // --- stage 1: sparse TransformerConv (t1) ---
  gemm(stream, nodes_bf, wq1b, t1_bq, nullptr, q1b, NTOT, D1, FIN, FIN,0, 1,FIN,0, D1,0, 1);
  gemm(stream, nodes_bf, wk1b, t1_bk, nullptr, k1b, NTOT, D1, FIN, FIN,0, 1,FIN,0, D1,0, 1);
  gemm(stream, nodes_bf, wv1b, t1_bv, nullptr, v1b, NTOT, D1, FIN, FIN,0, 1,FIN,0, D1,0, 1);
  gemm(stream, nodes_bf, ws1b, t1_bs, x1, nullptr, NTOT, D1, FIN, FIN,0, 1,FIN,0, D1,0, 1);
  FILLU(menc1, ENC_NEGINF, NTOT * 3);
  FILLF(z1, 0.f, NTOT * 3);
  k_t1_alpha<<<blks((long)NEDGE * 3), 256, 0, stream>>>(q1b, k1b, src, dst, alpha, menc1);
  k_t1_exp<<<blks((long)NEDGE * 3), 256, 0, stream>>>(dst, menc1, alpha, z1);
  k_t1_msg<<<blks((long)NEDGE * D1), 256, 0, stream>>>(alpha, z1, v1b, src, dst, x1);

  // --- stage 2: ASAPool 1 ---
  k_xmax_init<<<blks((long)NTOT * D1), 256, 0, stream>>>(x1, xme, NTOT * D1);
  k_xmax_edge<<<blks((long)NEDGE * D1), 256, 0, stream>>>(x1, src, dst, xme);
  k_enc_to_bf<<<blks((long)NTOT * D1), 256, 0, stream>>>(xme, xmaxb, NTOT * D1);
  gemm(stream, xmaxb, p1lwb, p1_lb, xq1, nullptr, NTOT, D1, D1, D1,0, 1,D1,0, D1,0, 1);
  k_matvec<<<blks(NTOT), 256, 0, stream>>>(xq1, p1_aw, nullptr, tqv, NTOT, D1);
  k_matvec<<<blks(NTOT), 256, 0, stream>>>(x1, p1_aw + D1, nullptr, tjv, NTOT, D1);
  FILLU(menc2, ENC_NEGINF, NTOT);
  FILLF(z2, 0.f, NTOT);
  k_p1_score<<<blks(NEE), 256, 0, stream>>>(src, dst, tqv, tjv, p1_ab, score, menc2);
  k_p1_exp<<<blks(NEE), 256, 0, stream>>>(src, dst, menc2, score, z2);
  k_p1_norm<<<blks(NEE), 256, 0, stream>>>(src, dst, z2, score);
  FILLF(xp1, 0.f, NTOT * D1);
  k_p1_xp<<<blks((long)NEE * D1), 256, 0, stream>>>(src, dst, score, x1, xp1);
  k_matvec<<<blks(NTOT), 256, 0, stream>>>(xp1, p1_le1w, p1_le1b, a1, NTOT, D1);
  k_matvec<<<blks(NTOT), 256, 0, stream>>>(xp1, p1_le2w, nullptr, bb1, NTOT, D1);
  FILLF(agg1, 0.f, NTOT);
  k_p1_agg<<<blks(NEE), 256, 0, stream>>>(src, dst, a1, bb1, agg1);
  k_fit<<<blks(NTOT), 256, 0, stream>>>(xp1, p1_le3w, p1_le3b, agg1, fit1, NTOT, D1);
  k_topk<<<NGR, NPGN, 0, stream>>>(fit1, NPGN, KP1, idx1, fsel1);
  k_gather_xsel1<<<blks((long)NGR * KP1 * D1), 256, 0, stream>>>(xp1, idx1, fsel1, xsel1b);
  FILLF(Sbuf, 0.f, NGR * NPGN * NPGN);
  FILLF(Abuf, 0.f, NGR * NPGN * NPGN);
  k_p1_SA<<<blks(NEE), 256, 0, stream>>>(src, dst, score, Sbuf, Abuf);
  k_gather_sselT<<<blks((long)NGR * KP1 * NPGN), 256, 0, stream>>>(Sbuf, idx1, sselT);
  CVT(Abuf, Abf, NGR * NPGN * NPGN);
  // T[i,l] = sum_j A[i,j]*Ssel[j,l]; B = sselT (k-contiguous)
  gemm(stream, Abf, sselT, nullptr, nullptr, Tbf, NPGN, KP1, NPGN,
       NPGN,(long)NPGN*NPGN, 1,NPGN,(long)KP1*NPGN, KP1,(long)NPGN*KP1, NGR);
  // A2[k,l] = sum_i Ssel[i,k]*T[i,l]; A = sselT (row-major), B = Tbf (n-contig)
  gemm(stream, sselT, Tbf, nullptr, A2, nullptr, KP1, KP1, NPGN,
       NPGN,(long)KP1*NPGN, KP1,1,(long)NPGN*KP1, KP1,(long)KP1*KP1, NGR);
  k_adj<<<blks((long)NGR * KP1 * KP1), 256, 0, stream>>>(A2, adj);

  // --- stage 3: dense TransformerConv (t2) ---
  gemm(stream, xsel1b, wq2b, t2_bq, nullptr, q2b, NGR*KP1, D2, D1, D1,0, 1,D1,0, D2,0, 1);
  gemm(stream, xsel1b, wk2b, t2_bk, nullptr, k2b, NGR*KP1, D2, D1, D1,0, 1,D1,0, D2,0, 1);
  gemm(stream, xsel1b, wv2b, t2_bv, nullptr, v2b, NGR*KP1, D2, D1, D1,0, 1,D1,0, D2,0, 1);
  gemm(stream, xsel1b, ws2b, t2_bs, x2, nullptr, NGR*KP1, D2, D1, D1,0, 1,D1,0, D2,0, 1);
  k_attn_logits_t2<<<dim3(KP1/16, 1, NGR*2), 128, 0, stream>>>(q2b, k2b, logits);
  k_t2_softmax<<<blks((long)2 * NGR * KP1), 256, 0, stream>>>(adj, logits);
  CVT(logits, attnb, NGR * 2 * KP1 * KP1);
  k_attn_apply_t2<<<dim3(HCD/16, 1, NGR*2), 128, 0, stream>>>(attnb, v2b, x2);
  CVT(x2, x2b, NGR * KP1 * D2);

  // --- stage 4: ASAPool 2 ---
  k_p2_xmax<<<blks((long)NGR * KP1 * D2), 256, 0, stream>>>(x2, adj, xmax2b);
  gemm(stream, xmax2b, p2lwb, p2_lb, xq2, nullptr, NGR*KP1, D2, D2, D2,0, 1,D2,0, D2,0, 1);
  k_matvec<<<blks(NGR*KP1), 256, 0, stream>>>(xq2, p2_aw, nullptr, si, NGR*KP1, D2);
  k_matvec<<<blks(NGR*KP1), 256, 0, stream>>>(x2, p2_aw + D2, nullptr, tj2, NGR*KP1, D2);
  k_p2_attn<<<blks(NGR*KP1), 256, 0, stream>>>(si, tj2, adj, p2_ab, attn2);
  CVT(attn2, attn2b, NGR * KP1 * KP1);
  // xp2 = attn2 @ x2  [b: 64x64 @ 64x256]
  gemm(stream, attn2b, x2b, nullptr, xp2, nullptr, KP1, D2, KP1,
       KP1,(long)KP1*KP1, D2,1,(long)KP1*D2, D2,(long)KP1*D2, NGR);
  k_matvec<<<blks(NGR*KP1), 256, 0, stream>>>(xp2, p2_le1w, p2_le1b, a2, NGR*KP1, D2);
  k_matvec<<<blks(NGR*KP1), 256, 0, stream>>>(xp2, p2_le2w, nullptr, bb2, NGR*KP1, D2);
  k_p2_agg<<<blks(NGR*KP1), 256, 0, stream>>>(adj, a2, bb2, agg2);
  k_fit<<<blks(NGR*KP1), 256, 0, stream>>>(xp2, p2_le3w, p2_le3b, agg2, fit2, NGR*KP1, D2);
  k_topk<<<NGR, KP1, 0, stream>>>(fit2, KP1, KP2, idx2, fsel2);
  k_pool_mean<<<blks((long)NGR * D2), 256, 0, stream>>>(xp2, idx2, fsel2, gmean);

  // --- stage 5: head MLP ---
  k_head1<<<blks((long)NGR * HCD), 256, 0, stream>>>(gmean, expv, depth, b1_w, b1_b, hbuf);
  k_head2<<<blks((long)NGR * NEV), 256, 0, stream>>>(hbuf, b2_w, b2_b, out);

#undef CVT
#undef CVTT
#undef FILLF
#undef FILLU
}
